// LinearAttention_22557168238737
// MI455X (gfx1250) — matrix operand order repaired and checked by reading, not compiled, not run on hardware
//
#include <hip/hip_runtime.h>

typedef __bf16 bf16;
typedef __attribute__((ext_vector_type(16))) __bf16 v16bf;
typedef __attribute__((ext_vector_type(8)))  float  v8f;
typedef __attribute__((ext_vector_type(4)))  __bf16 v4bf;
typedef __attribute__((ext_vector_type(4)))  unsigned v4u;
typedef __attribute__((ext_vector_type(8)))  int      v8i;
typedef __attribute__((ext_vector_type(4)))  int      v4i;

#define NH   36
#define DD   32
#define CC   1152            // == NH*DD
#define BB   4
#define NN   8192
#define MM   (BB*NN)         // 32768 rows
#define QKVN (3*CC)          // 3456

struct U16B2 { uint4 a, b; };           // 32 bytes
struct U4x8  { unsigned x[8]; };        // 32 bytes

// ---------------------------------------------------------------------------
// WMMA helpers (wave32, v_wmma_f32_16x16x32_bf16)
// ---------------------------------------------------------------------------
__device__ __forceinline__ v8f wmma_bf16(v16bf a, v16bf b, v8f c) {
    return __builtin_amdgcn_wmma_f32_16x16x32_bf16(
        /*neg_a=*/false, a, /*neg_b=*/false, b,
        /*c_mod=*/(short)0, c, /*reuse_a=*/false, /*reuse_b=*/false);
}

// A fragment (16x32 bf16, MxK). `row` points at M=(lane&15) row, K-contiguous.
// lanes 0-15 hold K = {0..15}; lanes 16-31 hold K = {16..31};
// each packed VGPR pair of K's is a contiguous u32 load.
__device__ __forceinline__ v16bf load_frag_a_kcontig(const bf16* row, int g) {
    U4x8 w;
#pragma unroll
    for (int j = 0; j < 8; ++j) {
        int k = 16 * g + 2 * j;
        w.x[j] = *reinterpret_cast<const unsigned*>(row + k);
    }
    return __builtin_bit_cast(v16bf, w);
}

// B fragment (32x16 bf16, KxN) from an N-major/K-contiguous tile.
// `p` points at column N=(lane&15)'s K-row, offset by 16*(lane>>4).
__device__ __forceinline__ v16bf load_frag_b_kcontig(const bf16* p) {
    U16B2 u;
    u.a = *reinterpret_cast<const uint4*>(p);
    u.b = *reinterpret_cast<const uint4*>(p + 8);
    return __builtin_bit_cast(v16bf, u);
}

// ---------------------------------------------------------------------------
// TDM: async DMA of a 2-D bf16 tile (32 K-elems x 128 rows) global -> LDS.
// D# per CDNA5 ISA §8.3/8.4. pad: +4 DWORDs every 16 DWORDs => LDS row pitch
// becomes 64B + 16B = 80B = LDK(40) bf16 elements, matching fragment loaders.
// Issued by one wave; completion via s_wait_tensorcnt + workgroup barrier.
// This toolchain exposes the 6-arg builtin: (v4u, v8i, v4i, v4i, v8i, cpol);
// groups 2/3 (+extra group) are zero for a 2-D tensor (tile_dim2..4 = 0).
// ---------------------------------------------------------------------------
#define LDK 40

__device__ __forceinline__ void tdm_load_tile(const bf16* gsrc, unsigned lds_off,
                                              unsigned tdim0, unsigned tdim1,
                                              unsigned row_stride) {
    unsigned long long ga = (unsigned long long)(uintptr_t)gsrc;
    v4u g0;
    g0[0] = 1u;                                   // count=1, user desc
    g0[1] = lds_off;                              // lds_addr (bytes)
    g0[2] = (unsigned)ga;                         // global_addr[31:0]
    g0[3] = (unsigned)(ga >> 32) | (2u << 30);    // global_addr[56:32] | type=2
    v8i g1;
    g1[0] = (int)((1u << 16)        // data_size = 2 bytes
                | (1u << 20)        // pad_enable
                | (3u << 22)        // pad_interval: 16 DWORDs
                | (3u << 25));      // pad_amount:   4 DWORDs
    g1[1] = (int)((tdim0 & 0xFFFFu) << 16);                    // tensor_dim0 lo
    g1[2] = (int)((tdim0 >> 16) | ((tdim1 & 0xFFFFu) << 16));  // dim0 hi | dim1 lo
    g1[3] = (int)((tdim1 >> 16) | (32u << 16));                // dim1 hi | tile_dim0=32
    g1[4] = (int)(128u);                                       // tile_dim1=128, tile_dim2=0
    g1[5] = (int)row_stride;                                   // tensor_dim0_stride lo32
    g1[6] = 0;                                                 // stride hi | dim1_stride lo
    g1[7] = 0;
    v4i z4 = {0, 0, 0, 0};
    v8i z8 = {0, 0, 0, 0, 0, 0, 0, 0};
    __builtin_amdgcn_tensor_load_to_lds(g0, g1, z4, z4, z8, 0);
}

// ---------------------------------------------------------------------------
// Kernel 0a: fp32 -> bf16 elementwise (4 per thread)
// ---------------------------------------------------------------------------
__global__ __launch_bounds__(256) void k_cvt_f32_bf16(const float* __restrict__ in,
                                                      bf16* __restrict__ out, size_t n4) {
    size_t i = (size_t)blockIdx.x * blockDim.x + threadIdx.x;
    if (i >= n4) return;
    float4 v = reinterpret_cast<const float4*>(in)[i];
    v4bf o;
    o[0] = (bf16)v.x; o[1] = (bf16)v.y; o[2] = (bf16)v.z; o[3] = (bf16)v.w;
    *reinterpret_cast<v4bf*>(out + i * 4) = o;
}

// ---------------------------------------------------------------------------
// Kernel 0b: transpose + convert.  in: R x C fp32 row-major -> out: C x R bf16
// ---------------------------------------------------------------------------
__global__ __launch_bounds__(256) void k_transpose_cvt(const float* __restrict__ in,
                                                       bf16* __restrict__ out,
                                                       int R, int C) {
    __shared__ float tile[32][33];
    int c0 = blockIdx.x * 32, r0 = blockIdx.y * 32;
    int tx = threadIdx.x & 31, ty = threadIdx.x >> 5;   // ty: 0..7
#pragma unroll
    for (int i = ty; i < 32; i += 8)
        tile[i][tx] = in[(size_t)(r0 + i) * C + c0 + tx];
    __syncthreads();
#pragma unroll
    for (int i = ty; i < 32; i += 8)
        out[(size_t)(c0 + i) * R + r0 + tx] = (bf16)tile[tx][i];
}

// ---------------------------------------------------------------------------
// Kernel 1/4: tiled bf16 GEMM, D(M,N) = A(M,K) * Bt(N,K)^T
// Block tile 128x128, K-step 32, 8 waves as 4(M) x 2(N); 8 WMMAs / wave / step.
// Staging: double-buffered TDM (tensor_load_to_lds) issued by wave 0,
// overlapped with WMMA compute on the other buffer.
// EPI==0: qkv epilogue (ReLU on q,k; split bf16 stores)
// EPI==1: out epilogue (add bias, fp32 store)
// ---------------------------------------------------------------------------
template <int EPI>
__global__ __launch_bounds__(256, 2)
void k_gemm_bf16(const bf16* __restrict__ A, const bf16* __restrict__ Bt,
                 int M, int N, int K,
                 bf16* __restrict__ qo, bf16* __restrict__ ko, bf16* __restrict__ vo,
                 float* __restrict__ out, const float* __restrict__ bias) {
    __shared__ bf16 lds_a[2][128 * LDK];
    __shared__ bf16 lds_b[2][128 * LDK];

    const int tid  = threadIdx.x;
    const int wid  = tid >> 5;
    const int lane = tid & 31;
    const int g    = lane >> 4;     // half-wave select
    const int ln   = lane & 15;
    const int wm   = wid >> 1;      // 0..3 -> 32 rows each
    const int wn   = wid & 1;       // 0..1 -> 64 cols each
    const int gm0  = blockIdx.x * 128;
    const int gn0  = blockIdx.y * 128;

    const bf16* Ab = A  + (size_t)gm0 * K;
    const bf16* Bb = Bt + (size_t)gn0 * K;
    const unsigned la[2] = { (unsigned)(uintptr_t)&lds_a[0][0],
                             (unsigned)(uintptr_t)&lds_a[1][0] };
    const unsigned lb[2] = { (unsigned)(uintptr_t)&lds_b[0][0],
                             (unsigned)(uintptr_t)&lds_b[1][0] };

    v8f acc[2][4] = {};

    // prologue: DMA first K-tile into buffer 0
    if (wid == 0) {
        tdm_load_tile(Ab, la[0], (unsigned)K, (unsigned)(M - gm0), (unsigned)K);
        tdm_load_tile(Bb, lb[0], (unsigned)K, (unsigned)(N - gn0), (unsigned)K);
        __builtin_amdgcn_s_wait_tensorcnt(0);
    }
    __syncthreads();

    int cur = 0;
    for (int k0 = 0; k0 < K; k0 += 32) {
        // async: DMA next K-tile into the other buffer while we compute
        if (wid == 0 && (k0 + 32) < K) {
            tdm_load_tile(Ab + k0 + 32, la[cur ^ 1], (unsigned)K, (unsigned)(M - gm0), (unsigned)K);
            tdm_load_tile(Bb + k0 + 32, lb[cur ^ 1], (unsigned)K, (unsigned)(N - gn0), (unsigned)K);
        }

        v16bf af[2], bf_[4];
#pragma unroll
        for (int i = 0; i < 2; ++i)
            af[i] = load_frag_a_kcontig(&lds_a[cur][(wm * 32 + i * 16 + ln) * LDK], g);
#pragma unroll
        for (int j = 0; j < 4; ++j)
            bf_[j] = load_frag_b_kcontig(&lds_b[cur][(wn * 64 + j * 16 + ln) * LDK + 16 * g]);
#pragma unroll
        for (int i = 0; i < 2; ++i)
#pragma unroll
            for (int j = 0; j < 4; ++j)
                acc[i][j] = wmma_bf16(af[i], bf_[j], acc[i][j]);

        if (wid == 0) __builtin_amdgcn_s_wait_tensorcnt(0);
        __syncthreads();
        cur ^= 1;
    }

    // ---- epilogue: D layout = col (lane&15), row (vgpr + 8*(lane>>4)) ----
#pragma unroll
    for (int i = 0; i < 2; ++i) {
        int rowbase = gm0 + wm * 32 + i * 16 + g * 8;
#pragma unroll
        for (int j = 0; j < 4; ++j) {
            int colbase = gn0 + wn * 64 + j * 16;     // 16-wide tile, never
            int seg = colbase / CC;                   // straddles a CC boundary
            int csub = colbase - seg * CC + ln;
            bf16* dst = (EPI == 0) ? ((seg == 0) ? qo : ((seg == 1) ? ko : vo)) : nullptr;
#pragma unroll
            for (int r = 0; r < 8; ++r) {
                float v = acc[i][j][r];
                size_t row = (size_t)(rowbase + r);
                if (EPI == 0) {
                    if (seg < 2 && v < 0.f) v = 0.f;   // ReLU on q,k
                    dst[row * CC + csub] = (bf16)v;
                } else {
                    int col = colbase + ln;
                    out[row * (size_t)N + col] = v + bias[col];
                }
            }
        }
    }
}

// ---------------------------------------------------------------------------
// Kernel 2: per (b,h) kv[32,32] = sum_n k[n,:]^T v[n,:]  and  k_sum[32]
// 144 blocks, 8 waves; each wave owns a 1024-row slice, WMMA over k^T tiles
// staged in wave-private LDS, then cross-wave LDS reduction.
// ---------------------------------------------------------------------------
__global__ __launch_bounds__(256, 2)
void k_kv_ksum(const bf16* __restrict__ kb, const bf16* __restrict__ vb,
               bf16* __restrict__ kvb, float* __restrict__ ksum) {
    __shared__ bf16  kst[8][32 * 32];
    __shared__ bf16  vst[8][32 * 32];
    __shared__ float red[8 * 1024];
    __shared__ float ksr[8][32];

    const int bh = blockIdx.x;
    const int b  = bh / NH, h = bh % NH;
    const int tid = threadIdx.x, wid = tid >> 5, lane = tid & 31;
    const int g = lane >> 4, ln = lane & 15;

    v8f   acc[2][2] = {};
    float ks = 0.f;

    for (int n0 = wid * 1024; n0 < wid * 1024 + 1024; n0 += 32) {
        // stage 32(n) x 32(d) bf16 blocks of k and v (coalesced 16B loads)
        size_t grow = ((size_t)(b * NN + n0 + lane)) * CC + h * DD;
        const uint4* sk = reinterpret_cast<const uint4*>(kb + grow);
        const uint4* sv = reinterpret_cast<const uint4*>(vb + grow);
        uint4* dk = reinterpret_cast<uint4*>(&kst[wid][lane * 32]);
        uint4* dv = reinterpret_cast<uint4*>(&vst[wid][lane * 32]);
#pragma unroll
        for (int c = 0; c < 4; ++c) { dk[c] = sk[c]; dv[c] = sv[c]; }

        // k_sum partial: lane == d
#pragma unroll
        for (int n = 0; n < 32; ++n) ks += (float)kst[wid][n * 32 + lane];

        // B fragments from v (K = n dimension)
        v16bf bbf[2];
#pragma unroll
        for (int et = 0; et < 2; ++et) {
#pragma unroll
            for (int e = 0; e < 16; ++e)
                bbf[et][e] = vst[wid][(e + 16 * g) * 32 + et * 16 + ln];
        }
        // A = k^T fragments: A[m=d, kk=n] = k[n, d]
#pragma unroll
        for (int dt = 0; dt < 2; ++dt) {
            v16bf a;
#pragma unroll
            for (int e = 0; e < 16; ++e) {
                int kk = 16 * g + e;
                a[e] = kst[wid][kk * 32 + dt * 16 + ln];
            }
#pragma unroll
            for (int et = 0; et < 2; ++et)
                acc[dt][et] = wmma_bf16(a, bbf[et], acc[dt][et]);
        }
    }

    // dump per-wave 32x32 accumulators: red[wid][d*32+e]
#pragma unroll
    for (int dt = 0; dt < 2; ++dt)
#pragma unroll
        for (int et = 0; et < 2; ++et)
#pragma unroll
            for (int i = 0; i < 8; ++i) {
                int d = dt * 16 + i + 8 * g, e = et * 16 + ln;
                red[wid * 1024 + d * 32 + e] = acc[dt][et][i];
            }
    ksr[wid][lane] = ks;
    __syncthreads();

    // cross-wave reduction -> kv bf16 (layout d-major 32x32)
#pragma unroll
    for (int i = 0; i < 4; ++i) {
        int idx = tid + 256 * i;
        float s = 0.f;
#pragma unroll
        for (int w = 0; w < 8; ++w) s += red[w * 1024 + idx];
        kvb[(size_t)bh * 1024 + idx] = (bf16)s;
    }
    if (wid == 0) {
        float s = 0.f;
#pragma unroll
        for (int w = 0; w < 8; ++w) s += ksr[w][lane];
        ksum[bh * 32 + lane] = s;
    }
}

// ---------------------------------------------------------------------------
// Kernel 3: attn[n, h*32+e] = (q[n,h,:] @ kv[h]) / (q[n,h,:].k_sum[h] + 1e-6)
// grid = B*H*(N/128); 8 waves x 16 rows; 1 A-frag + 2 B-frags -> 2 WMMAs/wave
// ---------------------------------------------------------------------------
__global__ __launch_bounds__(256, 4)
void k_attn(const bf16* __restrict__ qb, const bf16* __restrict__ kvb,
            const float* __restrict__ ksum, bf16* __restrict__ attn) {
    __shared__ bf16  kv_s[1024];
    __shared__ float denom[128];

    int gid = blockIdx.x;
    int nt = gid & 63;            // 64 row-tiles of 128
    int bh = gid >> 6;
    int b = bh / NH, h = bh % NH;
    int n0 = nt * 128;
    const int tid = threadIdx.x, wid = tid >> 5, lane = tid & 31;
    const int g = lane >> 4, ln = lane & 15;

    if (tid < 128)   // stage kv (2KB) coalesced
        *reinterpret_cast<uint4*>(&kv_s[tid * 8]) =
            *reinterpret_cast<const uint4*>(&kvb[(size_t)bh * 1024 + tid * 8]);
    if (tid < 128) { // denom per row
        const bf16* qrow = qb + ((size_t)(b * NN + n0 + tid)) * CC + h * DD;
        float s = 0.f;
#pragma unroll
        for (int d = 0; d < 32; ++d) s += (float)qrow[d] * ksum[bh * 32 + d];
        denom[tid] = s + 1e-6f;
    }
    __syncthreads();

    // A fragment straight from global q (K-contiguous 32 d's per row)
    const bf16* qrow = qb + ((size_t)(b * NN + n0 + wid * 16 + ln)) * CC + h * DD;
    v16bf a = load_frag_a_kcontig(qrow, g);

    v8f acc[2] = {};
#pragma unroll
    for (int et = 0; et < 2; ++et) {
        v16bf bb;
#pragma unroll
        for (int e = 0; e < 16; ++e)
            bb[e] = kv_s[(e + 16 * g) * 32 + et * 16 + ln];
        acc[et] = wmma_bf16(a, bb, acc[et]);
    }

#pragma unroll
    for (int et = 0; et < 2; ++et)
#pragma unroll
        for (int i = 0; i < 8; ++i) {
            int rl = wid * 16 + i + 8 * g;
            float v = acc[et][i] / denom[rl];
            attn[((size_t)(b * NN + n0 + rl)) * CC + h * DD + et * 16 + ln] = (bf16)v;
        }
}

// ---------------------------------------------------------------------------
// Host-side launcher
// ---------------------------------------------------------------------------
extern "C" void kernel_launch(void* const* d_in, const int* in_sizes, int n_in,
                              void* d_out, int out_size, void* d_ws, size_t ws_size,
                              hipStream_t stream) {
    const float* x     = (const float*)d_in[0];   // (B,N,C)
    const float* W_qkv = (const float*)d_in[1];   // (C, 3*H*D)
    const float* W_out = (const float*)d_in[2];   // (H*D, C)
    const float* b_out = (const float*)d_in[3];   // (C,)
    float* out = (float*)d_out;

    // workspace carve-out (256B aligned)
    size_t off = 0;
    auto alloc = [&](size_t bytes) {
        void* p = (char*)d_ws + off;
        off += (bytes + 255) & ~(size_t)255;
        return p;
    };
    bf16*  xb   = (bf16*)alloc((size_t)MM * CC * 2);     // x bf16; later reused for attn
    bf16*  WqT  = (bf16*)alloc((size_t)QKVN * CC * 2);   // W_qkv^T  (3456 x 1152)
    bf16*  WoT  = (bf16*)alloc((size_t)CC * CC * 2);     // W_out^T  (1152 x 1152)
    bf16*  qb   = (bf16*)alloc((size_t)MM * CC * 2);
    bf16*  kb   = (bf16*)alloc((size_t)MM * CC * 2);
    bf16*  vb   = (bf16*)alloc((size_t)MM * CC * 2);
    bf16*  kvb  = (bf16*)alloc((size_t)BB * NH * 1024 * 2);
    float* ksum = (float*)alloc((size_t)BB * NH * 32 * 4);
    bf16*  attn = xb;   // alias: x bf16 dead after GEMM1

    // 0) convert + transpose
    {
        size_t n4 = (size_t)MM * CC / 4;
        k_cvt_f32_bf16<<<(unsigned)((n4 + 255) / 256), 256, 0, stream>>>(x, xb, n4);
        k_transpose_cvt<<<dim3(QKVN / 32, CC / 32), 256, 0, stream>>>(W_qkv, WqT, CC, QKVN);
        k_transpose_cvt<<<dim3(CC / 32, CC / 32), 256, 0, stream>>>(W_out, WoT, CC, CC);
    }
    // 1) qkv projection + ReLU split   (32768 x 3456 x 1152)
    k_gemm_bf16<0><<<dim3(MM / 128, QKVN / 128), 256, 0, stream>>>(
        xb, WqT, MM, QKVN, CC, qb, kb, vb, nullptr, nullptr);
    // 2) kv = k^T v, k_sum            (144 blocks)
    k_kv_ksum<<<BB * NH, 256, 0, stream>>>(kb, vb, kvb, ksum);
    // 3) attn = (q @ kv) / denom      (B*H*64 blocks)
    k_attn<<<BB * NH * (NN / 128), 256, 0, stream>>>(qb, kvb, ksum, attn);
    // 4) out = attn @ W_out + b_out   (32768 x 1152 x 1152)
    k_gemm_bf16<1><<<dim3(MM / 128, CC / 128), 256, 0, stream>>>(
        attn, WoT, MM, CC, CC, nullptr, nullptr, nullptr, out, b_out);
}
